// TransformerAttentionLayer_83124797047409
// MI455X (gfx1250) — matrix operand order repaired and checked by reading, not compiled, not run on hardware
//
#include <hip/hip_runtime.h>

typedef __attribute__((ext_vector_type(16))) __bf16 v16bf;
typedef __attribute__((ext_vector_type(8)))  float  v8f;

#define BATCH 8
#define CH    64
#define IH    128
#define IW    256

// ---------------- fragment helpers (wave32 WMMA bf16 layouts) ----------------
union FragU { v16bf v; uint4 u[2]; };
union AccU  { v8f v; float f[8]; float4 q[2]; };

// A-matrix 16x32 bf16: lane<16 -> row m0+l, K = k0+{0..15}
//                      lane>=16 -> row m0+(l&15), K = k0+{16..31}
__device__ __forceinline__ v16bf ldsFragA(const __bf16* base, int row0, int stride,
                                          int k0, int lane) {
  const int l = lane & 15, half = lane >> 4;
  const __bf16* p = base + (size_t)(row0 + l) * stride + k0 + half * 16;
  FragU r;
  r.u[0] = *(const uint4*)(p);        // 8 bf16
  r.u[1] = *(const uint4*)(p + 8);    // 8 bf16, +16B
  return r.v;
}

// B-matrix 32x16 bf16 from [n][k] row-major LDS: lane<16 -> col n0+l, K=k0+{0..15}
//                                                lane>=16 -> col n0+(l&15), K=k0+{16..31}
__device__ __forceinline__ v16bf ldsFragB(const __bf16* base, int n0, int stride,
                                          int k0, int lane) {
  const int l = lane & 15, half = lane >> 4;
  const __bf16* p = base + (size_t)(n0 + l) * stride + k0 + half * 16;
  FragU r;
  r.u[0] = *(const uint4*)(p);
  r.u[1] = *(const uint4*)(p + 8);
  return r.v;
}

// B fragment from a per-lane row pointer (used by conv with tap-shifted pixels)
__device__ __forceinline__ v16bf ldsFragBrow(const __bf16* rowp, int k0, int half) {
  const __bf16* p = rowp + k0 + half * 16;
  FragU r;
  r.u[0] = *(const uint4*)(p);
  r.u[1] = *(const uint4*)(p + 8);
  return r.v;
}

__device__ __forceinline__ v8f wmma_bf16(v16bf a, v16bf b, v8f c) {
  return __builtin_amdgcn_wmma_f32_16x16x32_bf16(false, a, false, b, (short)0, c,
                                                 false, false);
}

// Low 32 bits of a flat shared-aperture address are the LDS byte offset
// (ISA 10.2).
__device__ __forceinline__ unsigned lds_addr32(const void* p) {
  return (unsigned)(uintptr_t)p;
}

// async global->LDS copy of 16B per lane, GVS addressing (ASYNCcnt-tracked)
__device__ __forceinline__ void async_copy_b128(unsigned lds_byte_addr,
                                                unsigned gl_byte_off,
                                                const void* gl_base) {
  asm volatile("global_load_async_to_lds_b128 %0, %1, %2"
               :
               : "v"(lds_byte_addr), "v"(gl_byte_off), "s"(gl_base)
               : "memory");
}

// ======================= conv3x3 + BN + ReLU (implicit GEMM) =================
struct ConvParams {
  const float* x[3];
  const float* W[3];
  const float* bias[3];
  const float* gamma[3];
  const float* beta[3];
  const float* mean[3];
  const float* var[3];
};

// block = (b, 8x32 pixel tile, which conv). 256 threads = 8 waves.
// LDS: Wlds[9][64][64] bf16, xt[(8+2)*(32+2)][64] bf16, invs/shifts[64] f32.
#define CONV_LDS_BYTES (9*64*64*2 + 340*64*2 + 2*64*4)

__global__ __launch_bounds__(256)
void conv_bn_relu_wmma(ConvParams cp, __bf16* outq, __bf16* outk, __bf16* outv) {
  extern __shared__ char smemraw[];
  __bf16* Wlds = (__bf16*)smemraw;           // [tap][co][ci]
  __bf16* xt   = Wlds + 9 * 64 * 64;         // [padded pixel][ci]
  float*  invs = (float*)(xt + 340 * 64);
  float*  shfs = invs + 64;

  const int which = blockIdx.z;
  const int b  = blockIdx.y;
  const int ht = blockIdx.x >> 3, wt = blockIdx.x & 7;
  const int h0 = ht * 8, w0 = wt * 32;
  const int tid  = threadIdx.x;
  const int wave = tid >> 5, lane = tid & 31;

  const float* xin = cp.x[which];
  const float* Wg  = cp.W[which];
  __bf16* outp = (which == 0) ? outq : ((which == 1) ? outk : outv);

  // warm GL2 for this block's input tile rows (global_prefetch)
  {
    int ph = tid % 10;                       // 10 padded rows
    int gh = h0 - 1 + ph;
    if (gh >= 0 && gh < IH)
      __builtin_prefetch(&xin[(((size_t)b * CH + (tid / 10) % CH) * IH + gh) * IW + w0],
                         0, 0);
  }

  // stage weights: global [co][ci][kh][kw] f32 -> LDS [tap][co][ci] bf16
  for (int idx = tid; idx < 9 * 64 * 64; idx += 256) {
    int co = idx / 576, rem = idx - co * 576;
    int ci = rem / 9,  tap = rem - ci * 9;
    Wlds[(tap * 64 + co) * 64 + ci] = (__bf16)Wg[idx];
  }
  // BN affine: y = conv*inv + (bias*inv + beta - mean*inv)
  if (tid < 64) {
    float inv = cp.gamma[which][tid] * rsqrtf(cp.var[which][tid] + 1e-5f);
    invs[tid] = inv;
    shfs[tid] = cp.bias[which][tid] * inv + cp.beta[which][tid]
              - cp.mean[which][tid] * inv;
  }
  // stage padded input tile, transposed to [pixel][ci] bf16 (zero halo)
  for (int idx = tid; idx < 340 * 64; idx += 256) {
    int ci = idx / 340, pix = idx - ci * 340;
    int ph = pix / 34,  pw  = pix - ph * 34;
    int gh = h0 - 1 + ph, gw = w0 - 1 + pw;
    float v = 0.f;
    if (gh >= 0 && gh < IH && gw >= 0 && gw < IW)
      v = xin[(((size_t)b * CH + ci) * IH + gh) * IW + gw];
    xt[pix * 64 + ci] = (__bf16)v;
  }
  __syncthreads();

  // wave owns 4 M-tiles (all 64 co) x 2 N-tiles (32 pixels)
  v8f zero = {};
  v8f acc[4][2];
#pragma unroll
  for (int tm = 0; tm < 4; ++tm)
#pragma unroll
    for (int tn = 0; tn < 2; ++tn) acc[tm][tn] = zero;

  int prow[2], pcol[2];
#pragma unroll
  for (int tn = 0; tn < 2; ++tn) {
    int p = 16 * (wave * 2 + tn) + (lane & 15);
    prow[tn] = p >> 5;
    pcol[tn] = p & 31;
  }

  for (int tap = 0; tap < 9; ++tap) {
    int kh = tap / 3, kw = tap - kh * 3;
    const __bf16* Wt = Wlds + tap * 64 * 64;
#pragma unroll
    for (int ks = 0; ks < 2; ++ks) {
      int k0 = ks * 32;
      v16bf a0 = ldsFragA(Wt,  0, 64, k0, lane);
      v16bf a1 = ldsFragA(Wt, 16, 64, k0, lane);
      v16bf a2 = ldsFragA(Wt, 32, 64, k0, lane);
      v16bf a3 = ldsFragA(Wt, 48, 64, k0, lane);
#pragma unroll
      for (int tn = 0; tn < 2; ++tn) {
        const __bf16* rp = xt + ((prow[tn] + kh) * 34 + (pcol[tn] + kw)) * 64;
        v16bf bb = ldsFragBrow(rp, k0, lane >> 4);
        acc[0][tn] = wmma_bf16(a0, bb, acc[0][tn]);
        acc[1][tn] = wmma_bf16(a1, bb, acc[1][tn]);
        acc[2][tn] = wmma_bf16(a2, bb, acc[2][tn]);
        acc[3][tn] = wmma_bf16(a3, bb, acc[3][tn]);
      }
    }
  }

  // epilogue: BN + ReLU, store bf16 as [b][h][w][c] (16B contiguous per lane)
  const int nlo = lane & 15, hn = lane >> 4;
#pragma unroll
  for (int tm = 0; tm < 4; ++tm) {
    int co = 16 * tm + hn * 8;
    float iv[8], sh[8];
#pragma unroll
    for (int j = 0; j < 8; ++j) { iv[j] = invs[co + j]; sh[j] = shfs[co + j]; }
#pragma unroll
    for (int tn = 0; tn < 2; ++tn) {
      int p  = 16 * (wave * 2 + tn) + nlo;
      int gh = h0 + (p >> 5), gw = w0 + (p & 31);
      union { __bf16 hh[8]; uint4 u; } pk;
#pragma unroll
      for (int j = 0; j < 8; ++j) {
        float y = acc[tm][tn][j] * iv[j] + sh[j];
        pk.hh[j] = (__bf16)fmaxf(y, 0.f);
      }
      *(uint4*)(outp + ((((size_t)b * IH + gh) * IW + gw) * CH + co)) = pk.u;
    }
  }
}

// ============== per-(b,h) attention: S = K^T Q, softmax, O = P V^T ==========
// LDS: Qt[256][64], Kt[256][64], Vc[64][256] bf16, P[8 waves][16][256] bf16
#define ATTN_LDS_BYTES ((3 * IW * CH + 8 * 16 * IW) * 2)

__global__ __launch_bounds__(256)
void attention_wmma(const __bf16* __restrict__ qbf, const __bf16* __restrict__ kbf,
                    const __bf16* __restrict__ vbf, float* __restrict__ out) {
  extern __shared__ char smemraw[];
  __bf16* Qt = (__bf16*)smemraw;          // [v][c]
  __bf16* Kt = Qt + IW * CH;              // [w][c]
  __bf16* Vc = Kt + IW * CH;              // [c][v]  (transposed on stage)
  __bf16* Pl = Vc + CH * IW;              // [wave][16][256]

  const int h = blockIdx.x, b = blockIdx.y;
  const int tid = threadIdx.x, wave = tid >> 5, lane = tid & 31;
  const size_t base = ((size_t)b * IH + h) * (size_t)(IW * CH);

  // ---- Qt / Kt: straight 32KB copies through the async global->LDS engine ----
  {
    unsigned qdst = lds_addr32(Qt) + (unsigned)tid * 16u;
    unsigned kdst = lds_addr32(Kt) + (unsigned)tid * 16u;
    unsigned goff = (unsigned)tid * 16u;
    const void* qsrc = (const void*)(qbf + base);
    const void* ksrc = (const void*)(kbf + base);
#pragma unroll
    for (int i = 0; i < 8; ++i) {   // 8 passes * 256 thr * 16B = 32KB each
      async_copy_b128(qdst, goff, qsrc);
      async_copy_b128(kdst, goff, ksrc);
      qdst += 4096u; kdst += 4096u; goff += 4096u;
    }
  }
  // ---- V: transpose [w][c] -> [c][v] (must round-trip through VGPRs) ----
  for (int i = tid; i < IW * CH; i += 256) {
    int w = i >> 6, c = i & 63;
    Vc[c * IW + w] = vbf[base + i];
  }
  asm volatile("s_wait_asynccnt 0x0" ::: "memory");  // barrier ignores ASYNCcnt
  __syncthreads();

  const float scale = 1.f / (64.f + 1e-7f);
  __bf16* Pw = Pl + wave * 16 * IW;
  const int nlo = lane & 15, hn = lane >> 4;
  v8f zero = {};

  for (int sidx = 0; sidx < 2; ++sidx) {
    const int strip = wave + sidx * 8;     // 16 strips of 16 rows cover w=0..255
    const int w0 = strip * 16;

    // ---- GEMM1: S[w0:w0+16][0:256] = K^T Q ----
    v8f acc[16];
#pragma unroll
    for (int t = 0; t < 16; ++t) acc[t] = zero;
#pragma unroll
    for (int ks = 0; ks < 2; ++ks) {
      int k0 = ks * 32;
      v16bf a = ldsFragA(Kt, w0, CH, k0, lane);   // A[m=w][k=c] = Kt[w][c]
#pragma unroll
      for (int tv = 0; tv < 16; ++tv) {
        v16bf bb = ldsFragB(Qt, 16 * tv, CH, k0, lane); // B[k=c][n=v] = Qt[v][c]
        acc[tv] = wmma_bf16(a, bb, acc[tv]);
      }
    }

    // ---- softmax over v (rows match C-fragment layout: half-wave = 8 rows) ----
#pragma unroll
    for (int j = 0; j < 8; ++j) {
      float m = -3.4e38f;
#pragma unroll
      for (int t = 0; t < 16; ++t) m = fmaxf(m, acc[t][j]);
      m = fmaxf(m, __shfl_xor(m, 1));
      m = fmaxf(m, __shfl_xor(m, 2));
      m = fmaxf(m, __shfl_xor(m, 4));
      m = fmaxf(m, __shfl_xor(m, 8));
      float s = 0.f;
#pragma unroll
      for (int t = 0; t < 16; ++t) {
        float p = __expf((acc[t][j] - m) * scale);
        acc[t][j] = p;
        s += p;
      }
      s += __shfl_xor(s, 1);
      s += __shfl_xor(s, 2);
      s += __shfl_xor(s, 4);
      s += __shfl_xor(s, 8);
      float rinv = 1.f / s;
#pragma unroll
      for (int t = 0; t < 16; ++t) acc[t][j] *= rinv;
    }

    // ---- spill P strip to LDS (row w-local = hn*8 + j, col = 16*t + nlo) ----
#pragma unroll
    for (int t = 0; t < 16; ++t)
#pragma unroll
      for (int j = 0; j < 8; ++j)
        Pw[(hn * 8 + j) * IW + 16 * t + nlo] = (__bf16)acc[t][j];

    asm volatile("s_wait_dscnt 0x0" ::: "memory");  // same-wave LDS RAW fence

    // ---- GEMM2: O[16 x 64] = P (16x256) * V^T ----
    v8f oac[4];
#pragma unroll
    for (int tc = 0; tc < 4; ++tc) oac[tc] = zero;
#pragma unroll
    for (int ks = 0; ks < 8; ++ks) {
      int k0 = ks * 32;
      v16bf a = ldsFragA(Pw, 0, IW, k0, lane);        // A[m=w][k=v] = Pw[w][v]
#pragma unroll
      for (int tc = 0; tc < 4; ++tc) {
        v16bf bb = ldsFragB(Vc, 16 * tc, IW, k0, lane); // B[k=v][n=c] = Vc[c][v]
        oac[tc] = wmma_bf16(a, bb, oac[tc]);
      }
    }

    // ---- store O to [B][C][H][W] f32; j indexes consecutive w ----
#pragma unroll
    for (int tc = 0; tc < 4; ++tc) {
      int c = 16 * tc + nlo;
      int wrow = w0 + hn * 8;
      float* op = out + (((size_t)b * CH + c) * IH + h) * IW + wrow;
      AccU u;
      u.v = oac[tc];
      *(float4*)(op)     = u.q[0];
      *(float4*)(op + 4) = u.q[1];
    }
  }
}

// ============================== launch ======================================
extern "C" void kernel_launch(void* const* d_in, const int* in_sizes, int n_in,
                              void* d_out, int out_size, void* d_ws, size_t ws_size,
                              hipStream_t stream) {
  (void)in_sizes; (void)n_in; (void)out_size;

  const float* cur = (const float*)d_in[0];
  const float* oth = (const float*)d_in[1];

  ConvParams cp;
  cp.x[0] = cur; cp.x[1] = oth; cp.x[2] = oth;
  for (int i = 0; i < 3; ++i) {
    int o = 2 + 6 * i;
    cp.W[i]     = (const float*)d_in[o + 0];
    cp.bias[i]  = (const float*)d_in[o + 1];
    cp.gamma[i] = (const float*)d_in[o + 2];
    cp.beta[i]  = (const float*)d_in[o + 3];
    cp.mean[i]  = (const float*)d_in[o + 4];
    cp.var[i]   = (const float*)d_in[o + 5];
  }

  const size_t N = (size_t)BATCH * IH * IW * CH;   // 16.7M elems per tensor
  __bf16* qbf = (__bf16*)d_ws;                     // needs 3*N*2 = ~96 MB ws
  __bf16* kbf = qbf + N;
  __bf16* vbf = kbf + N;
  (void)ws_size;

  (void)hipFuncSetAttribute((const void*)conv_bn_relu_wmma,
                            hipFuncAttributeMaxDynamicSharedMemorySize,
                            CONV_LDS_BYTES);
  (void)hipFuncSetAttribute((const void*)attention_wmma,
                            hipFuncAttributeMaxDynamicSharedMemorySize,
                            ATTN_LDS_BYTES);

  // conv: 16 h-tiles * 8 w-tiles, B batches, 3 convs
  conv_bn_relu_wmma<<<dim3(128, BATCH, 3), 256, CONV_LDS_BYTES, stream>>>(
      cp, qbf, kbf, vbf);

  // attention: one block per (h, b)
  attention_wmma<<<dim3(IH, BATCH), 256, ATTN_LDS_BYTES, stream>>>(
      qbf, kbf, vbf, (float*)d_out);
}
